// SGSLanguageModel_20890720928528
// MI455X (gfx1250) — matrix operand order repaired and checked by reading, not compiled, not run on hardware
//
#include <hip/hip_runtime.h>
#include <math.h>

// ---------------------------------------------------------------------------
// SGS language model for MI455X (gfx1250, wave32, WMMA).
// All GEMMs: v_wmma_f32_16x16x32_bf16, 128x128x32 block tiles, 8 waves,
// wave tile 32x64 (8 WMMA / k-step), double-buffered LDS fed by
// global_load_async_to_lds_b128 when the toolchain exposes it.
// ---------------------------------------------------------------------------

typedef __bf16 bf16;
typedef __attribute__((ext_vector_type(16))) __bf16 v16bf;
typedef __attribute__((ext_vector_type(8)))  __bf16 v8bf;
typedef __attribute__((ext_vector_type(8)))  float  v8f;
typedef __attribute__((ext_vector_type(4)))  int    v4i;

namespace {
constexpr int V_  = 32000;
constexpr int DS  = 128;     // d_s
constexpr int DF  = 1024;    // d_f
constexpr int NH  = 4;       // heads
constexpr int NP  = 3;       // passes
constexpr int B_  = 2;
constexpr int L_  = 2048;
constexpr int R_  = B_ * L_;     // 4096 flattened rows
constexpr int CTXD = 2 * DF;     // 2048
constexpr int HD   = 4 * DF;     // 4096
}

enum { EPI_NONE = 0, EPI_TANH = 1, EPI_GELU = 2, EPI_KERNEL = 3 };

union Frag { v16bf v; v8bf h[2]; bf16 e[16]; };

#if __has_builtin(__builtin_amdgcn_global_load_async_to_lds_b128) && \
    __has_builtin(__builtin_amdgcn_s_wait_asynccnt)
#define USE_ASYNC 1
#else
#define USE_ASYNC 0
#endif

#if USE_ASYNC
#define AS_GLOBAL __attribute__((address_space(1)))
#define AS_LDS    __attribute__((address_space(3)))
static __device__ __forceinline__ void async_cp16(const bf16* g, bf16* l) {
    __builtin_amdgcn_global_load_async_to_lds_b128(
        (AS_GLOBAL v4i*)g,
        (AS_LDS    v4i*)l, 0, 0);
}
#endif

// ---------------------------------------------------------------------------
// WMMA GEMM:  C[M,N] = epi(A[M,K] @ B[N,K]^T + bias) (+ resid)
// Batched over blockIdx.z with element strides sA/sB/sC/sCvec.
// ---------------------------------------------------------------------------
template<int EPI>
__global__ __launch_bounds__(256) void gemm_wmma(
    const bf16* __restrict__ A, const bf16* __restrict__ B,
    float* __restrict__ Cf, bf16* __restrict__ Cb,
    const float* __restrict__ bias, const float* __restrict__ resid,
    const float* __restrict__ cvec, const float* __restrict__ ltau,
    int M, int N, int K, int ldc,
    long sA, long sB, long sC, long sCvec)
{
    constexpr int BM = 128, BN = 128, BK = 32, LDT = BK + 8; // 80B rows, 16B aligned
    __shared__ __align__(16) bf16 sA_[2][BM][LDT];
    __shared__ __align__(16) bf16 sB_[2][BN][LDT];

    const int z = blockIdx.z;
    A += z * sA;
    B += z * sB;
    if (Cf)    Cf    += z * sC;
    if (Cb)    Cb    += z * sC;
    if (resid) resid += z * sC;
    if (cvec)  cvec  += z * sCvec;

    const int tid  = threadIdx.x;
    const int wave = tid >> 5, lane = tid & 31;
    const int half = lane >> 4, lm = lane & 15;
    const int m0 = blockIdx.y * BM, n0 = blockIdx.x * BN;
    const int wm = (wave & 3) * 32;     // wave M sub-tile (32 rows)
    const int wn = (wave >> 2) * 64;    // wave N sub-tile (64 cols)

    // cooperative staging: thread covers rows rr and rr+64, 16B each
    const int rr = tid >> 2, cc = (tid & 3) * 8;

    v8f acc[2][4];
#pragma unroll
    for (int i = 0; i < 2; ++i)
#pragma unroll
        for (int j = 0; j < 4; ++j) acc[i][j] = (v8f){};

    const int nk = K / BK;

#if USE_ASYNC
    auto issue = [&](int t, int buf) {
        const int k0 = t * BK;
        async_cp16(&A[(long)(m0 + rr) * K + k0 + cc],      &sA_[buf][rr][cc]);
        async_cp16(&A[(long)(m0 + rr + 64) * K + k0 + cc], &sA_[buf][rr + 64][cc]);
        async_cp16(&B[(long)(n0 + rr) * K + k0 + cc],      &sB_[buf][rr][cc]);
        async_cp16(&B[(long)(n0 + rr + 64) * K + k0 + cc], &sB_[buf][rr + 64][cc]);
    };
    issue(0, 0);
#else
    {   // stage tile 0 synchronously
        v8bf t0 = *(const v8bf*)&A[(long)(m0 + rr) * K + cc];
        v8bf t1 = *(const v8bf*)&A[(long)(m0 + rr + 64) * K + cc];
        v8bf t2 = *(const v8bf*)&B[(long)(n0 + rr) * K + cc];
        v8bf t3 = *(const v8bf*)&B[(long)(n0 + rr + 64) * K + cc];
        *(v8bf*)&sA_[0][rr][cc]      = t0;
        *(v8bf*)&sA_[0][rr + 64][cc] = t1;
        *(v8bf*)&sB_[0][rr][cc]      = t2;
        *(v8bf*)&sB_[0][rr + 64][cc] = t3;
    }
#endif

    for (int t = 0; t < nk; ++t) {
        const int buf = t & 1;
#if USE_ASYNC
        __builtin_amdgcn_s_wait_asynccnt(0);
        __syncthreads();
        if (t + 1 < nk) issue(t + 1, buf ^ 1);
#else
        __syncthreads();
        v8bf t0, t1, t2, t3;
        const bool pf = (t + 1 < nk);
        if (pf) {
            const int k0 = (t + 1) * BK;
            t0 = *(const v8bf*)&A[(long)(m0 + rr) * K + k0 + cc];
            t1 = *(const v8bf*)&A[(long)(m0 + rr + 64) * K + k0 + cc];
            t2 = *(const v8bf*)&B[(long)(n0 + rr) * K + k0 + cc];
            t3 = *(const v8bf*)&B[(long)(n0 + rr + 64) * K + k0 + cc];
        }
#endif
        // fragments per ISA 7.12.2 (16-bit A layout; B mirrors with n=lane%16)
        Frag a0, a1, b[4];
        const bf16* ap0 = &sA_[buf][wm + lm][half * 8];
        a0.h[0] = *(const v8bf*)ap0;
        a0.h[1] = *(const v8bf*)(ap0 + 16);
        const bf16* ap1 = &sA_[buf][wm + 16 + lm][half * 8];
        a1.h[0] = *(const v8bf*)ap1;
        a1.h[1] = *(const v8bf*)(ap1 + 16);
#pragma unroll
        for (int j = 0; j < 4; ++j) {
            const bf16* bp = &sB_[buf][wn + j * 16 + lm][half * 8];
            b[j].h[0] = *(const v8bf*)bp;
            b[j].h[1] = *(const v8bf*)(bp + 16);
        }
#pragma unroll
        for (int j = 0; j < 4; ++j) {
            acc[0][j] = __builtin_amdgcn_wmma_f32_16x16x32_bf16(
                false, a0.v, false, b[j].v, (short)0, acc[0][j], false, false);
            acc[1][j] = __builtin_amdgcn_wmma_f32_16x16x32_bf16(
                false, a1.v, false, b[j].v, (short)0, acc[1][j], false, false);
        }
#if !USE_ASYNC
        if (pf) {
            *(v8bf*)&sA_[buf ^ 1][rr][cc]      = t0;
            *(v8bf*)&sA_[buf ^ 1][rr + 64][cc] = t1;
            *(v8bf*)&sB_[buf ^ 1][rr][cc]      = t2;
            *(v8bf*)&sB_[buf ^ 1][rr + 64][cc] = t3;
        }
#endif
    }

    float kscale = 0.f;
    if constexpr (EPI == EPI_KERNEL) kscale = -0.5f * __expf(-ltau[0]); // -0.5/tau

#pragma unroll
    for (int j = 0; j < 4; ++j) {
        const int n = n0 + wn + j * 16 + lm;
        const float bb = bias ? bias[n] : 0.f;
        float cv = 0.f;
        if constexpr (EPI == EPI_KERNEL) cv = cvec[n];
#pragma unroll
        for (int i = 0; i < 2; ++i) {
            const int mr = m0 + wm + i * 16 + half * 8;
#pragma unroll
            for (int r = 0; r < 8; ++r) {
                const long idx = (long)(mr + r) * ldc + n;
                float x = acc[i][j][r] + bb;
                if constexpr (EPI == EPI_TANH)   x = tanhf(x);
                if constexpr (EPI == EPI_GELU)   x = 0.5f * x * (1.f + erff(x * 0.70710678f));
                if constexpr (EPI == EPI_KERNEL) x = __expf((x + cv) * kscale);
                if (resid) x += resid[idx];
                if (Cf) Cf[idx] = x;
                if (Cb) Cb[idx] = (bf16)x;
            }
        }
    }
}

// ---------------------------------------------------------------------------
// f32 -> bf16 conversion (grid-strided)
// ---------------------------------------------------------------------------
__global__ void k_cvt(const float* __restrict__ s, bf16* __restrict__ d, long n) {
    long i  = (long)blockIdx.x * blockDim.x + threadIdx.x;
    long st = (long)gridDim.x * blockDim.x;
    for (; i < n; i += st) d[i] = (bf16)s[i];
}

// 32x32 LDS-tiled transpose: src [L_, DF] -> dst [DF, L_] per batch z.
__global__ __launch_bounds__(256) void k_transpose(const bf16* __restrict__ src,
                                                   bf16* __restrict__ dst) {
    __shared__ bf16 tile[32][34];
    const int b = blockIdx.z;
    src += (long)b * L_ * DF;
    dst += (long)b * L_ * DF;
    const int c0 = blockIdx.x * 32;             // feature dim
    const int r0 = blockIdx.y * 32;             // sequence dim
    const int tx = threadIdx.x & 31, ty = threadIdx.x >> 5;
#pragma unroll
    for (int i = 0; i < 32; i += 8)
        tile[ty + i][tx] = src[(long)(r0 + ty + i) * DF + c0 + tx];
    __syncthreads();
#pragma unroll
    for (int i = 0; i < 32; i += 8)
        dst[(long)(c0 + ty + i) * L_ + r0 + tx] = tile[tx][ty + i];
}

// ---------------------------------------------------------------------------
// Embedding gather + positional add + alpha/iv init. One block per (b,l).
// ---------------------------------------------------------------------------
__global__ __launch_bounds__(256) void k_embed(
    const int* __restrict__ ids, const float* __restrict__ tok_mu,
    const float* __restrict__ tok_lv, const float* __restrict__ tok_ra,
    const float* __restrict__ tok_feat, const float* __restrict__ pos_mu,
    float* __restrict__ mu, float* __restrict__ iv, float* __restrict__ alpha,
    float* __restrict__ feat, bf16* __restrict__ featb)
{
    const int row = blockIdx.x, l = row % L_, tid = threadIdx.x;
    const int id = ids[row];
    if (tid < DS) {
        float m = tok_mu[(long)id * DS + tid] + pos_mu[(long)l * DS + tid];
        mu[(long)row * DS + tid] = m;
        iv[(long)row * DS + tid] = __expf(-tok_lv[(long)id * DS + tid]);
    }
    for (int c = tid; c < DF; c += 256) {
        float f = tok_feat[(long)id * DF + c];
        feat[(long)row * DF + c]  = f;
        featb[(long)row * DF + c] = (bf16)f;
    }
    if (tid == 0) alpha[row] = 1.f / (1.f + __expf(-tok_ra[id]));
}

// ---------------------------------------------------------------------------
// Per-pass prep: mu_bf16, u = [iv ; mu*iv] (bf16), mu2s = sum(mu^2*iv).
// ---------------------------------------------------------------------------
__global__ __launch_bounds__(128) void k_prep(
    const float* __restrict__ mu, const float* __restrict__ iv,
    bf16* __restrict__ mub, bf16* __restrict__ u, float* __restrict__ mu2s)
{
    const int row = blockIdx.x, tid = threadIdx.x;
    __shared__ float red[128];
    float m = mu[(long)row * DS + tid];
    float v = iv[(long)row * DS + tid];
    mub[(long)row * DS + tid]        = (bf16)m;
    u[(long)row * 2 * DS + tid]      = (bf16)v;
    u[(long)row * 2 * DS + DS + tid] = (bf16)(m * v);
    red[tid] = m * m * v;
    __syncthreads();
    for (int s = 64; s > 0; s >>= 1) { if (tid < s) red[tid] += red[tid + s]; __syncthreads(); }
    if (tid == 0) mu2s[row] = red[0];
}

// v = [q^2 ; -2q] (bf16) so that v.u + mu2s = Mahalanobis distance
__global__ __launch_bounds__(128) void k_vbuild(const float* __restrict__ q,
                                                bf16* __restrict__ vb) {
    const int row = blockIdx.x, tid = threadIdx.x;
    float x = q[(long)row * DS + tid];
    vb[(long)row * 2 * DS + tid]      = (bf16)(x * x);
    vb[(long)row * 2 * DS + DS + tid] = (bf16)(-2.f * x);
}

// ---------------------------------------------------------------------------
// Causal render weights: per (b,t) row, eff_a=clip(alpha_j*K_tj*mask),
// W_tj = eff_a * exp(prefix_{j'<j} log1p(-eff_a)).  Block-wide scan.
// ---------------------------------------------------------------------------
__global__ __launch_bounds__(256) void k_scan(
    const float* __restrict__ Kmat, const float* __restrict__ alpha,
    bf16* __restrict__ W)
{
    const int t = blockIdx.x, b = blockIdx.y;
    const float* krow = Kmat + ((long)b * L_ + t) * L_;
    const float* arow = alpha + (long)b * L_;
    bf16* wrow = W + ((long)b * L_ + t) * L_;
    const int tid = threadIdx.x, lane = tid & 31, wid = tid >> 5;
    __shared__ float wsum[8];
    __shared__ float wpre[9];
    float carry = 0.f;
    for (int j0 = 0; j0 < L_; j0 += 256) {
        const int j = j0 + tid;
        float a = (j <= t) ? arow[j] * krow[j] : 0.f;
        a = fminf(fmaxf(a, 0.f), 0.999999f);
        const float x = log1pf(-a);
        float v = x;
        for (int d = 1; d < 32; d <<= 1) {
            float y = __shfl_up(v, d);
            if (lane >= d) v += y;
        }
        if (lane == 31) wsum[wid] = v;
        __syncthreads();
        if (tid == 0) {
            float s = 0.f;
            for (int i = 0; i < 8; ++i) { wpre[i] = s; s += wsum[i]; }
            wpre[8] = s;
        }
        __syncthreads();
        const float excl = v + wpre[wid] - x;
        wrow[j] = (bf16)(a * __expf(carry + excl));
        carry += wpre[8];
        __syncthreads();
    }
}

// ---------------------------------------------------------------------------
// ctx = [features ; meaning]; write ctx_bf16, LN(ctx)_bf16; alpha *= sigmoid(ag).
// ---------------------------------------------------------------------------
__global__ __launch_bounds__(256) void k_ctx(
    const float* __restrict__ feat, const float* __restrict__ meaning,
    const float* __restrict__ ln_g, const float* __restrict__ ln_b,
    const float* __restrict__ ag_w, const float* __restrict__ ag_b,
    float* __restrict__ alpha, bf16* __restrict__ ctxb, bf16* __restrict__ ctxlnb)
{
    const int row = blockIdx.x, tid = threadIdx.x;
    __shared__ float red[256];
    __shared__ float stats[2];
    const float* f  = feat + (long)row * DF;
    const float* mg = meaning + (long)row * DF;
    float s = 0.f, s2 = 0.f, xs[8];
#pragma unroll
    for (int i = 0; i < 8; ++i) {
        const int c = tid + i * 256;
        float x = (c < DF) ? f[c] : mg[c - DF];
        xs[i] = x; s += x; s2 += x * x;
    }
    red[tid] = s; __syncthreads();
    for (int k = 128; k > 0; k >>= 1) { if (tid < k) red[tid] += red[tid + k]; __syncthreads(); }
    if (tid == 0) stats[0] = red[0] * (1.f / CTXD);
    __syncthreads();
    red[tid] = s2; __syncthreads();
    for (int k = 128; k > 0; k >>= 1) { if (tid < k) red[tid] += red[tid + k]; __syncthreads(); }
    if (tid == 0) {
        float m = stats[0];
        stats[1] = rsqrtf(red[0] * (1.f / CTXD) - m * m + 1e-5f);
    }
    __syncthreads();
    const float mean = stats[0], rstd = stats[1];
    float ad = 0.f;
#pragma unroll
    for (int i = 0; i < 8; ++i) {
        const int c = tid + i * 256;
        const float x = xs[i];
        ctxb[(long)row * CTXD + c]   = (bf16)x;
        ctxlnb[(long)row * CTXD + c] = (bf16)((x - mean) * rstd * ln_g[c] + ln_b[c]);
        ad += ag_w[c] * x;
    }
    red[tid] = ad; __syncthreads();
    for (int k = 128; k > 0; k >>= 1) { if (tid < k) red[tid] += red[tid + k]; __syncthreads(); }
    if (tid == 0) alpha[row] *= 1.f / (1.f + __expf(-(red[0] + ag_b[0])));
}

// Final LayerNorm of meaning -> bf16 for the LM-head GEMM.
__global__ __launch_bounds__(256) void k_lnf(
    const float* __restrict__ meaning, const float* __restrict__ g,
    const float* __restrict__ bb, bf16* __restrict__ outb)
{
    const int row = blockIdx.x, tid = threadIdx.x;
    __shared__ float red[256];
    __shared__ float stats[2];
    const float* mrow = meaning + (long)row * DF;
    float s = 0.f, s2 = 0.f, xs[4];
#pragma unroll
    for (int i = 0; i < 4; ++i) {
        float x = mrow[tid + i * 256];
        xs[i] = x; s += x; s2 += x * x;
    }
    red[tid] = s; __syncthreads();
    for (int k = 128; k > 0; k >>= 1) { if (tid < k) red[tid] += red[tid + k]; __syncthreads(); }
    if (tid == 0) stats[0] = red[0] * (1.f / DF);
    __syncthreads();
    red[tid] = s2; __syncthreads();
    for (int k = 128; k > 0; k >>= 1) { if (tid < k) red[tid] += red[tid + k]; __syncthreads(); }
    if (tid == 0) {
        float m = stats[0];
        stats[1] = rsqrtf(red[0] * (1.f / DF) - m * m + 1e-5f);
    }
    __syncthreads();
    const float m = stats[0], r = stats[1];
#pragma unroll
    for (int i = 0; i < 4; ++i) {
        const int c = tid + i * 256;
        outb[(long)row * DF + c] = (bf16)((xs[i] - m) * r * g[c] + bb[c]);
    }
}

// ---------------------------------------------------------------------------
// Host orchestration
// ---------------------------------------------------------------------------
extern "C" void kernel_launch(void* const* d_in, const int* in_sizes, int n_in,
                              void* d_out, int out_size, void* d_ws, size_t ws_size,
                              hipStream_t stream)
{
    (void)in_sizes; (void)n_in; (void)out_size; (void)ws_size;

    const int*   ids      = (const int*)  d_in[0];
    const float* tok_mu   = (const float*)d_in[1];
    const float* tok_lv   = (const float*)d_in[2];
    const float* tok_ra   = (const float*)d_in[3];
    const float* tok_feat = (const float*)d_in[4];
    const float* pos_mu   = (const float*)d_in[5];
    const float* log_tau  = (const float*)d_in[6];
    const float* qp_w     = (const float*)d_in[7];
    const float* qp_b     = (const float*)d_in[8];
    const float* hp_w     = (const float*)d_in[9];
    const float* hp_b     = (const float*)d_in[10];
    const float* mu_w     = (const float*)d_in[11];
    const float* mu_b     = (const float*)d_in[12];
    const float* ag_w     = (const float*)d_in[13];
    const float* ag_b     = (const float*)d_in[14];
    const float* ln_g     = (const float*)d_in[15];
    const float* ln_b     = (const float*)d_in[16];
    const float* ffn_w1   = (const float*)d_in[17];
    const float* ffn_b1   = (const float*)d_in[18];
    const float* ffn_w2   = (const float*)d_in[19];
    const float* ffn_b2   = (const float*)d_in[20];
    const float* lnf_g    = (const float*)d_in[21];
    const float* lnf_b    = (const float*)d_in[22];
    const float* lm_w     = (const float*)d_in[23];
    float* logits = (float*)d_out;

    // ---- bump allocator over d_ws ------------------------------------------
    char* wsp = (char*)d_ws;
    auto alloc = [&](size_t bytes) -> void* {
        void* r = (void*)wsp;
        wsp += (bytes + 255) & ~(size_t)255;
        return r;
    };
    float* mu      = (float*)alloc((size_t)R_ * DS * 4);
    float* iv      = (float*)alloc((size_t)R_ * DS * 4);
    float* alpha   = (float*)alloc((size_t)R_ * 4);
    float* feat    = (float*)alloc((size_t)R_ * DF * 4);
    bf16*  featb   = (bf16*) alloc((size_t)R_ * DF * 2);
    bf16*  featTb  = (bf16*) alloc((size_t)R_ * DF * 2);   // [B, DF, L]
    bf16*  mub     = (bf16*) alloc((size_t)R_ * DS * 2);
    bf16*  u       = (bf16*) alloc((size_t)R_ * 2 * DS * 2);
    float* mu2s    = (float*)alloc((size_t)R_ * 4);
    float* q       = (float*)alloc((size_t)R_ * DS * 4);
    bf16*  vb      = (bf16*) alloc((size_t)R_ * 2 * DS * 2);
    float* Kmat    = (float*)alloc((size_t)B_ * L_ * L_ * 4);
    bf16*  W       = (bf16*) alloc((size_t)B_ * L_ * L_ * 2);
    bf16*  headcatb= (bf16*) alloc((size_t)R_ * HD * 2);
    float* meaning = (float*)alloc((size_t)R_ * DF * 4);
    bf16*  ctxb    = (bf16*) alloc((size_t)R_ * CTXD * 2);
    bf16*  ctxlnb  = (bf16*) alloc((size_t)R_ * CTXD * 2);
    bf16*  h2b     = (bf16*) alloc((size_t)R_ * HD * 2);
    bf16*  lnfb    = (bf16*) alloc((size_t)R_ * DF * 2);
    bf16*  qpwb    = (bf16*) alloc((size_t)NH * DS * DS * 2);
    bf16*  hpwb    = (bf16*) alloc((size_t)DF * HD * 2);
    bf16*  muwb    = (bf16*) alloc((size_t)(NP - 1) * DS * CTXD * 2);
    bf16*  w1b     = (bf16*) alloc((size_t)(NP - 1) * HD * CTXD * 2);
    bf16*  w2b     = (bf16*) alloc((size_t)(NP - 1) * DF * HD * 2);
    bf16*  lmwb    = (bf16*) alloc((size_t)V_ * DF * 2);

    auto cvt = [&](const float* s, bf16* d, long n) {
        long blocks = (n + 255) / 256;
        if (blocks > 4096) blocks = 4096;
        k_cvt<<<dim3((unsigned)blocks), dim3(256), 0, stream>>>(s, d, n);
    };

    // embeddings + weight conversion
    k_embed<<<R_, 256, 0, stream>>>(ids, tok_mu, tok_lv, tok_ra, tok_feat, pos_mu,
                                    mu, iv, alpha, feat, featb);
    cvt(qp_w,   qpwb, (long)NH * DS * DS);
    cvt(hp_w,   hpwb, (long)DF * HD);
    cvt(mu_w,   muwb, (long)(NP - 1) * DS * CTXD);
    cvt(ffn_w1, w1b,  (long)(NP - 1) * HD * CTXD);
    cvt(ffn_w2, w2b,  (long)(NP - 1) * DF * HD);
    cvt(lm_w,   lmwb, (long)V_ * DF);

    for (int p = 0; p < NP; ++p) {
        k_prep<<<R_, 128, 0, stream>>>(mu, iv, mub, u, mu2s);
        // features^T for the render GEMM (features change each pass)
        k_transpose<<<dim3(DF / 32, L_ / 32, B_), 256, 0, stream>>>(featb, featTb);

        for (int h = 0; h < NH; ++h) {
            // q_h = mu @ qp_w[h]^T + qp_b[h]         [R_,128]
            gemm_wmma<EPI_NONE><<<dim3(DS / 128, R_ / 128, 1), 256, 0, stream>>>(
                mub, qpwb + (long)h * DS * DS, q, nullptr,
                qp_b + h * DS, nullptr, nullptr, nullptr,
                R_, DS, DS, DS, 0, 0, 0, 0);
            k_vbuild<<<R_, 128, 0, stream>>>(q, vb);

            // K_h[b] = exp(-0.5*(v@u^T + mu2s_j)/tau)   [B,L,L]
            gemm_wmma<EPI_KERNEL><<<dim3(L_ / 128, L_ / 128, B_), 256, 0, stream>>>(
                vb, u, Kmat, nullptr,
                nullptr, nullptr, mu2s, log_tau,
                L_, L_, 2 * DS, L_,
                (long)L_ * 2 * DS, (long)L_ * 2 * DS, (long)L_ * L_, (long)L_);

            // causal transmittance weights (bf16)
            k_scan<<<dim3(L_, B_), 256, 0, stream>>>(Kmat, alpha, W);

            // head output: W[b] @ featT[b]^T -> headcat[:, h*DF : (h+1)*DF]
            gemm_wmma<EPI_NONE><<<dim3(DF / 128, L_ / 128, B_), 256, 0, stream>>>(
                W, featTb, nullptr, headcatb + (long)h * DF,
                nullptr, nullptr, nullptr, nullptr,
                L_, DF, L_, HD,
                (long)L_ * L_, (long)L_ * DF, (long)L_ * HD, 0);
        }

        // meaning = headcat @ hp_w^T + hp_b
        gemm_wmma<EPI_NONE><<<dim3(DF / 128, R_ / 128, 1), 256, 0, stream>>>(
            headcatb, hpwb, meaning, nullptr,
            hp_b, nullptr, nullptr, nullptr,
            R_, DF, HD, DF, 0, 0, 0, 0);

        if (p < NP - 1) {
            k_ctx<<<R_, 256, 0, stream>>>(feat, meaning,
                ln_g + (long)p * CTXD, ln_b + (long)p * CTXD,
                ag_w + (long)p * CTXD, ag_b + p,
                alpha, ctxb, ctxlnb);

            // mu += tanh(ctx @ mu_w[p]^T + mu_b[p])
            gemm_wmma<EPI_TANH><<<dim3(DS / 128, R_ / 128, 1), 256, 0, stream>>>(
                ctxb, muwb + (long)p * DS * CTXD, mu, nullptr,
                mu_b + p * DS, mu, nullptr, nullptr,
                R_, DS, CTXD, DS, 0, 0, 0, 0);

            // h2 = gelu(LN(ctx) @ w1^T + b1)   (bf16 only — feeds next GEMM)
            gemm_wmma<EPI_GELU><<<dim3(HD / 128, R_ / 128, 1), 256, 0, stream>>>(
                ctxlnb, w1b + (long)p * HD * CTXD, nullptr, h2b,
                ffn_b1 + (long)p * HD, nullptr, nullptr, nullptr,
                R_, HD, CTXD, HD, 0, 0, 0, 0);

            // features += h2 @ w2^T + b2   (f32 master + bf16 copy)
            gemm_wmma<EPI_NONE><<<dim3(DF / 128, R_ / 128, 1), 256, 0, stream>>>(
                h2b, w2b + (long)p * DF * HD, feat, featb,
                ffn_b2 + (long)p * DF, feat, nullptr, nullptr,
                R_, DF, HD, DF, 0, 0, 0, 0);
        }
    }

    // logits = LN(meaning) @ lm_w^T
    k_lnf<<<R_, 256, 0, stream>>>(meaning, lnf_g, lnf_b, lnfb);
    gemm_wmma<EPI_NONE><<<dim3(V_ / 128, R_ / 128, 1), 256, 0, stream>>>(
        lnfb, lmwb, logits, nullptr,
        nullptr, nullptr, nullptr, nullptr,
        R_, V_, DF, V_, 0, 0, 0, 0);
}
